// CausalSelfAttention_35476429865354
// MI455X (gfx1250) — compile-verified, matrix operand order checked by reading, not run on hardware
//
#include <hip/hip_runtime.h>

// ---------------- problem constants ----------------
#define BATCH    4
#define SEQ      2048
#define DM       1024          // d_model
#define NH       16            // heads
#define HD       64            // head dim
#define C3       (3 * DM)      // 3072
#define MTOK     (BATCH * SEQ) // 8192

typedef __attribute__((ext_vector_type(16))) __bf16 v16bf;
typedef __attribute__((ext_vector_type(8)))  float  v8f;
typedef __attribute__((ext_vector_type(4)))  unsigned int v4u;
typedef __attribute__((ext_vector_type(8)))  int    v8i;
typedef __attribute__((ext_vector_type(4)))  int    v4i;

// K-index mapping for 16-bit A/B WMMA fragments (ISA 7.12.2):
//   elems 0..7  -> K = (lane<16 ? 0..7  : 8..15)
//   elems 8..15 -> K = (lane<16 ? 16..23: 24..31)
__device__ __forceinline__ int kmap(int j, int lane) {
    return (j & 7) + ((j >> 3) << 4) + ((lane >> 4) << 3);
}

// ---------------- Tensor Data Mover: 2D tile -> LDS ----------------
// Builds a D# (groups 0/1 per ISA 08_async_tensor §8) for a 2D bf16 tile:
//   tile_d0 elems per row (contiguous), tile_d1 rows, row stride in elems,
//   LDS padding: pad_interval code pi, pad_amount code pa (DWORD units).
__device__ __forceinline__ void tdm_load_2d(unsigned lds_addr, const void* gptr,
                                            int tile_d0, int tile_d1,
                                            int row_stride_elems,
                                            int pi, int pa) {
    unsigned long long ga = (unsigned long long)(uintptr_t)gptr;
    v4u g0;
    g0[0] = 1u;                                   // count=1 valid descriptor
    g0[1] = lds_addr;                             // bits 63:32 lds_addr
    g0[2] = (unsigned)(ga & 0xffffffffu);         // global_addr[31:0]
    g0[3] = (unsigned)((ga >> 32) & 0x1ffffffu)   // global_addr[56:32]
          | (2u << 30);                           // type=2 ("image")
    v8i g1;
    g1[0] = (1 << 16)                             // data_size: 2 bytes
          | (1 << 20)                             // pad_enable
          | ((pi & 7) << 22)                      // pad_interval
          | ((pa & 127) << 25);                   // pad_amount
    g1[1] = (tile_d0 & 0xffff) << 16;             // tensor_dim0[15:0] @ bits 63:48
    g1[2] = ((unsigned)tile_d0 >> 16)             // tensor_dim0[31:16]
          | ((tile_d1 & 0xffff) << 16);           // tensor_dim1[15:0]
    g1[3] = ((unsigned)tile_d1 >> 16)             // tensor_dim1[31:16]
          | ((tile_d0 & 0xffff) << 16);           // tile_dim0 @ bits 127:112
    g1[4] = (tile_d1 & 0xffff);                   // tile_dim1 (tile_dim2 = 0)
    g1[5] = row_stride_elems;                     // tensor_dim0_stride[31:0]
    g1[6] = 0;                                    // stride hi / dim1_stride lo
    g1[7] = 0;
    v4i z4 = {0, 0, 0, 0};
#if __clang_major__ >= 23
    v8i z8 = {0, 0, 0, 0, 0, 0, 0, 0};
    __builtin_amdgcn_tensor_load_to_lds(g0, g1, z4, z4, z8, 0);
#else
    __builtin_amdgcn_tensor_load_to_lds(g0, g1, z4, z4, 0);
#endif
}

// ---------------- DPP row reductions (16-lane rows, wave32) ----------------
// row_ror:n tree over n = 1,2,4,8 fully reduces each 16-lane row; pure VALU.
#define ROR_MAX(v, n)                                                          \
    {                                                                          \
        int _t = __builtin_amdgcn_update_dpp(0, __float_as_int(v),             \
                                             0x120 + (n), 0xf, 0xf, true);     \
        (v) = fmaxf((v), __int_as_float(_t));                                  \
    }
#define ROR_ADD(v, n)                                                          \
    {                                                                          \
        int _t = __builtin_amdgcn_update_dpp(0, __float_as_int(v),             \
                                             0x120 + (n), 0xf, 0xf, true);     \
        (v) = (v) + __int_as_float(_t);                                        \
    }
__device__ __forceinline__ float rowmax16(float v) {
    ROR_MAX(v, 1) ROR_MAX(v, 2) ROR_MAX(v, 4) ROR_MAX(v, 8)
    return v;
}
__device__ __forceinline__ float rowsum16(float v) {
    ROR_ADD(v, 1) ROR_ADD(v, 2) ROR_ADD(v, 4) ROR_ADD(v, 8)
    return v;
}

// ---------------- fp32 -> bf16 convert ----------------
__global__ void cvt_bf16_kernel(const float* __restrict__ in,
                                __bf16* __restrict__ out, int n) {
    int i = blockIdx.x * blockDim.x + threadIdx.x;
    int stride = gridDim.x * blockDim.x;
    for (; i < n; i += stride) out[i] = (__bf16)in[i];
}

// ---------------- GEMM: C[M,N] = A[M,K](bf16) @ B[K,N](bf16) + bias ------
// block tile 128x64, 256 threads = 8 waves in 4x2 grid, wave tile 32x32,
// K-step 32. Tiles staged by the Tensor Data Mover (wave 0 issues, all wait).
template <bool OUT_F32>
__global__ __launch_bounds__(256)
void gemm_bf16_kernel(const __bf16* __restrict__ A, const __bf16* __restrict__ B,
                      const float* __restrict__ bias,
                      float* __restrict__ Cf, __bf16* __restrict__ Cb,
                      int M, int N, int K) {
    const int tid  = threadIdx.x;
    const int lane = tid & 31;
    const int wave = tid >> 5;
    const int wm = wave >> 1;              // 0..3
    const int wn = wave & 1;               // 0..1
    const int blockM = blockIdx.y * 128;
    const int blockN = blockIdx.x * 64;

    __shared__ __align__(16) __bf16 Alds[128 * 48]; // 32-elem rows padded to 48
    __shared__ __align__(16) __bf16 Blds[32 * 72];  // 64-elem rows padded to 72

    v8f acc[2][2] = {};

    for (int k0 = 0; k0 < K; k0 += 32) {
        __syncthreads();                   // previous tiles fully consumed
        if (wave == 0) {
            // A tile 128x32: row=16 DW, pad 8 DW -> stride 48 elems (pi=3,pa=7)
            tdm_load_2d((unsigned)(uintptr_t)Alds,
                        A + (size_t)blockM * K + k0, 32, 128, K, 3, 7);
            // B tile 32x64: row=32 DW, pad 4 DW -> stride 72 elems (pi=4,pa=3)
            tdm_load_2d((unsigned)(uintptr_t)Blds,
                        B + (size_t)k0 * N + blockN, 64, 32, N, 4, 3);
            __builtin_amdgcn_s_wait_tensorcnt(0);
        }
        __syncthreads();

        v16bf afrag[2], bfrag[2];
#pragma unroll
        for (int mi = 0; mi < 2; ++mi) {
            int rowb = wm * 32 + mi * 16 + (lane & 15);
#pragma unroll
            for (int j = 0; j < 16; ++j)
                afrag[mi][j] = Alds[rowb * 48 + kmap(j, lane)];
        }
#pragma unroll
        for (int ni = 0; ni < 2; ++ni) {
            int colb = wn * 32 + ni * 16 + (lane & 15);
#pragma unroll
            for (int j = 0; j < 16; ++j)
                bfrag[ni][j] = Blds[kmap(j, lane) * 72 + colb];
        }
#pragma unroll
        for (int mi = 0; mi < 2; ++mi)
#pragma unroll
            for (int ni = 0; ni < 2; ++ni)
                acc[mi][ni] = __builtin_amdgcn_wmma_f32_16x16x32_bf16(
                    false, afrag[mi], false, bfrag[ni], (short)0,
                    acc[mi][ni], false, false);
    }

    // epilogue: C layout -> lane: M = r + 8*(lane>>4), N = lane&15
#pragma unroll
    for (int mi = 0; mi < 2; ++mi) {
#pragma unroll
        for (int ni = 0; ni < 2; ++ni) {
            int col = blockN + wn * 32 + ni * 16 + (lane & 15);
            float bv = bias ? bias[col] : 0.0f;
#pragma unroll
            for (int r = 0; r < 8; ++r) {
                int row = blockM + wm * 32 + mi * 16 + r + ((lane >> 4) << 3);
                float v = acc[mi][ni][r] + bv;
                if (OUT_F32) Cf[(size_t)row * N + col] = v;
                else         Cb[(size_t)row * N + col] = (__bf16)v;
            }
        }
    }
}

// ---------------- flash attention ----------------
// grid (SEQ/64, BATCH*NH), 128 threads = 4 waves; wave w handles 16 query rows.
// 32-key blocks staged by TDM; S via 2 WMMAs per 16-key subtile, DPP online
// softmax, P@V via 4 WMMAs.
__global__ __launch_bounds__(128)
void flash_attn_kernel(const __bf16* __restrict__ qkv, __bf16* __restrict__ y) {
    const int tid  = threadIdx.x;
    const int lane = tid & 31;
    const int w    = tid >> 5;              // wave 0..3
    const int q0   = blockIdx.x * 64;
    const int b    = blockIdx.y / NH;
    const int h    = blockIdx.y % NH;
    const size_t rowbase = (size_t)b * SEQ;

    __shared__ __align__(16) __bf16 Qs[64 * 72];
    __shared__ __align__(16) __bf16 Ks[32 * 72];
    __shared__ __align__(16) __bf16 Vs[32 * 72];
    __shared__ __bf16 Ps[4][16][36];        // per-wave P tile (16 rows x 32 keys)

    // stage Q tile 64x64 via TDM (row stride C3, pad -> LDS stride 72)
    if (w == 0) {
        tdm_load_2d((unsigned)(uintptr_t)Qs,
                    qkv + (rowbase + q0) * C3 + h * HD, 64, 64, C3, 4, 3);
        __builtin_amdgcn_s_wait_tensorcnt(0);
    }
    __syncthreads();

    // Q A-fragments for the two 32-wide K-dim chunks of head dim 64
    v16bf aQ[2];
#pragma unroll
    for (int c = 0; c < 2; ++c)
#pragma unroll
        for (int j = 0; j < 16; ++j)
            aQ[c][j] = Qs[(w * 16 + (lane & 15)) * 72 + c * 32 + kmap(j, lane)];

    float mrow[8], lrow[8];
    v8f O[4] = {};
#pragma unroll
    for (int r = 0; r < 8; ++r) { mrow[r] = -1e30f; lrow[r] = 0.0f; }
    const float scale = 0.125f;             // HD^-0.5 = 1/8

    for (int kb = 0; kb * 32 <= q0 + 63; ++kb) {
        __syncthreads();                    // all waves done with prev K/V tile
        if (w == 0) {
            const __bf16* kv = qkv + (rowbase + kb * 32) * C3 + h * HD;
            tdm_load_2d((unsigned)(uintptr_t)Ks, kv + DM,     64, 32, C3, 4, 3);
            tdm_load_2d((unsigned)(uintptr_t)Vs, kv + 2 * DM, 64, 32, C3, 4, 3);
            __builtin_amdgcn_s_wait_tensorcnt(0);
        }
        __syncthreads();

        // S = Q @ K^T : two 16x16 score tiles over 32 keys
        v8f s[2] = {};
#pragma unroll
        for (int nt = 0; nt < 2; ++nt) {
#pragma unroll
            for (int c = 0; c < 2; ++c) {
                v16bf bK;
#pragma unroll
                for (int j = 0; j < 16; ++j)
                    bK[j] = Ks[(nt * 16 + (lane & 15)) * 72 + c * 32 + kmap(j, lane)];
                s[nt] = __builtin_amdgcn_wmma_f32_16x16x32_bf16(
                    false, aQ[c], false, bK, (short)0, s[nt], false, false);
            }
        }

        // online softmax over the 32 keys; DPP row reductions (16-lane rows)
#pragma unroll
        for (int r = 0; r < 8; ++r) {
            int rowg = q0 + w * 16 + r + ((lane >> 4) << 3);
            float s0 = s[0][r] * scale;
            float s1 = s[1][r] * scale;
            int c0 = kb * 32 + (lane & 15);
            if (c0 > rowg)      s0 = -1e30f;
            if (c0 + 16 > rowg) s1 = -1e30f;
            float t  = rowmax16(fmaxf(s0, s1));
            float mn = fmaxf(mrow[r], t);
            float sc = __expf(mrow[r] - mn);
            float p0 = __expf(s0 - mn);
            float p1 = __expf(s1 - mn);
            float rs = rowsum16(p0 + p1);
            lrow[r] = lrow[r] * sc + rs;
            mrow[r] = mn;
#pragma unroll
            for (int dt = 0; dt < 4; ++dt) O[dt][r] *= sc;
            int prow = r + ((lane >> 4) << 3);
            Ps[w][prow][lane & 15]        = (__bf16)p0;
            Ps[w][prow][16 + (lane & 15)] = (__bf16)p1;
        }

        // O += P @ V  (K-dim = 32 keys, N = head dims in 4 tiles of 16)
        v16bf aP;
#pragma unroll
        for (int j = 0; j < 16; ++j)
            aP[j] = Ps[w][lane & 15][kmap(j, lane)];
#pragma unroll
        for (int dt = 0; dt < 4; ++dt) {
            v16bf bV;
#pragma unroll
            for (int j = 0; j < 16; ++j)
                bV[j] = Vs[kmap(j, lane) * 72 + dt * 16 + (lane & 15)];
            O[dt] = __builtin_amdgcn_wmma_f32_16x16x32_bf16(
                false, aP, false, bV, (short)0, O[dt], false, false);
        }
    }

    // write y (bf16, [B*T, DM] with head offset) normalized by row sums
#pragma unroll
    for (int dt = 0; dt < 4; ++dt) {
        int col = h * HD + dt * 16 + (lane & 15);
#pragma unroll
        for (int r = 0; r < 8; ++r) {
            int rowg = q0 + w * 16 + r + ((lane >> 4) << 3);
            y[(rowbase + rowg) * (size_t)DM + col] = (__bf16)(O[dt][r] / lrow[r]);
        }
    }
}

// ---------------- launcher ----------------
extern "C" void kernel_launch(void* const* d_in, const int* in_sizes, int n_in,
                              void* d_out, int out_size, void* d_ws, size_t ws_size,
                              hipStream_t stream) {
    const float* x    = (const float*)d_in[0];
    const float* Wqkv = (const float*)d_in[1];
    const float* bqkv = (const float*)d_in[2];
    const float* Wo   = (const float*)d_in[3];
    const float* bo   = (const float*)d_in[4];
    float* out = (float*)d_out;

    // workspace carve-up (bf16 staging + intermediates), 256B aligned
    char* ws = (char*)d_ws;
    size_t off = 0;
    auto carve = [&](size_t bytes) {
        char* p = ws + off;
        off = (off + bytes + 255) & ~(size_t)255;
        return p;
    };
    __bf16* xb    = (__bf16*)carve((size_t)MTOK * DM * 2);   // 16.8 MB
    __bf16* wqkvb = (__bf16*)carve((size_t)DM * C3 * 2);     //  6.3 MB
    __bf16* wob   = (__bf16*)carve((size_t)DM * DM * 2);     //  2.1 MB
    __bf16* qkvb  = (__bf16*)carve((size_t)MTOK * C3 * 2);   // 50.3 MB
    __bf16* yb    = (__bf16*)carve((size_t)MTOK * DM * 2);   // 16.8 MB

    // 1) convert fp32 inputs to bf16
    cvt_bf16_kernel<<<2048, 256, 0, stream>>>(x,    xb,    MTOK * DM);
    cvt_bf16_kernel<<<2048, 256, 0, stream>>>(Wqkv, wqkvb, DM * C3);
    cvt_bf16_kernel<<<1024, 256, 0, stream>>>(Wo,   wob,   DM * DM);

    // 2) qkv = x @ Wqkv + bqkv  -> bf16 [MTOK, 3072]
    gemm_bf16_kernel<false><<<dim3(C3 / 64, MTOK / 128), 256, 0, stream>>>(
        xb, wqkvb, bqkv, nullptr, qkvb, MTOK, C3, DM);

    // 3) flash attention -> yb bf16 [MTOK, DM]
    flash_attn_kernel<<<dim3(SEQ / 64, BATCH * NH), 128, 0, stream>>>(qkvb, yb);

    // 4) out = y @ Wo + bo -> fp32
    gemm_bf16_kernel<true><<<dim3(DM / 64, MTOK / 128), 256, 0, stream>>>(
        yb, wob, bo, out, nullptr, MTOK, DM, DM);
}
